// MultiHeadAttention_68410239091266
// MI455X (gfx1250) — matrix operand order repaired and checked by reading, not compiled, not run on hardware
//
#include <hip/hip_runtime.h>

#define BATCH 2
#define SEQ   2048
#define DM    768
#define NH    12
#define DH    64

typedef __attribute__((ext_vector_type(16))) _Float16 v16h;
typedef __attribute__((ext_vector_type(8)))  _Float16 v8h;
typedef __attribute__((ext_vector_type(8)))  float    v8f;

// C[m,n] = sum_k A[m,k] * B'[k,n] (+ bias[n]); fp32 in, f16 WMMA, fp32 accum.
// Block tile 128(M) x 64(N) x 32(K); 8 waves, each wave owns a 16x64 C strip
// (one A fragment feeds 4 WMMAs). LDS tiles are double-buffered and stored
// pre-permuted into WMMA fragment order (contiguous k = (lane>>4)*16 + j for A and B):
//   afrag[sub][lane][j]: A_tile[sub*16 + (lane&15)][ (lane>>4)*16 + j ]
//   bfrag[sub][lane][j]: B_tile[(lane>>4)*16 + j][ sub*16 + (lane&15) ]
// so fragment loads are contiguous ds_load_b128 pairs.
//
// BL=0: B stored [N,K] row-major (NT: X @ W^T, q @ k^T)
// BL=1: B stored [K,N] row-major (NN: weights @ v)
// CM=0: C row-major ldc, batch offset z*sC
// CM=1: head-split scatter  C[((b*NH+h)*SEQ+s)*DH + dd]   (-> [B,H,S,d])
// CM=2: head-merge          C[(b*SEQ+m)*DM + h*DH + n]    (-> [B,S,H*d])
template<int BL, int CM>
__global__ __launch_bounds__(256, 2)
void gemm_wmma_f16(const float* __restrict__ A, const float* __restrict__ B,
                   const float* __restrict__ bias, float* __restrict__ C,
                   int K, int lda, int ldb, int ldc,
                   long sA, long sB, long sC)
{
    // rows padded to 24 halfs (48B): 16B-aligned b128 rows, conflict-free banks
    __shared__ __align__(16) _Float16 afrag[2][8][32][24];   // 24 KB
    __shared__ __align__(16) _Float16 bfrag[2][4][32][24];   // 12 KB

    const int tid  = threadIdx.x;
    const int lane = tid & 31;
    const int wave = tid >> 5;                  // 0..7 : M sub-tile
    const int n0   = blockIdx.x * 64;
    const int m0   = blockIdx.y * 128;
    const float* Az = A + (long)blockIdx.z * sA;
    const float* Bz = B + (long)blockIdx.z * sB;

    // ---- staging coordinates ----
    // A: 128(M) x 32(K) = 4096 elems -> 16/thread as two 8-elem chunks
    //    (rows ar and ar+64, k = ac..ac+7)
    const int ar     = (tid * 8) >> 5;          // 0..63
    const int ac     = (tid * 8) & 31;          // 0,8,16,24
    const int a_sub  = ar >> 4;                 // 0..3 (second chunk: +4)
    const int a_lane = ((ac >> 4) << 4) + (ar & 15);
    const int a_j    = ac & 8;                  // 0 or 8
    // B: 2048 elems -> 8/thread
    // NT: n = br (0..63), k = bc..bc+7 ;  NN: k = br (0..31), n = bc..bc+7
    const int br = (BL == 0) ? ((tid * 8) >> 5) : ((tid * 8) >> 6);
    const int bc = (BL == 0) ? ((tid * 8) & 31) : ((tid * 8) & 63);
    const int bnt_sub  = br >> 4;
    const int bnt_lane = ((bc >> 4) << 4) + (br & 15);
    const int bnt_j    = bc & 8;
    const int bnn_sub  = bc >> 4;
    const int bnn_lane = ((br >> 4) << 4) + (bc & 15);
    const int bnn_j    = br & 15;

    float4 aR0a, aR0b, aR1a, aR1b, bRa, bRb;    // staged tile in registers

    auto load_regs = [&](int kk) {
        const float* p0 = Az + (long)(m0 + ar) * lda + (kk + ac);
        aR0a = *(const float4*)p0;
        aR0b = *(const float4*)(p0 + 4);
        const float* p1 = Az + (long)(m0 + ar + 64) * lda + (kk + ac);
        aR1a = *(const float4*)p1;
        aR1b = *(const float4*)(p1 + 4);
        const float* pb = (BL == 0)
            ? (Bz + (long)(n0 + br) * ldb + (kk + bc))
            : (Bz + (long)(kk + br) * ldb + (n0 + bc));
        bRa = *(const float4*)pb;
        bRb = *(const float4*)(pb + 4);
        if (kk + 32 < K) {                       // global_prefetch_b8
            __builtin_prefetch(p0 + 32, 0, 3);
            __builtin_prefetch(p1 + 32, 0, 3);
            __builtin_prefetch((BL == 0) ? (pb + 32) : (pb + (long)32 * ldb), 0, 3);
        }
    };

    auto cvt8 = [](const float4& f0, const float4& f1) -> v8h {
        v8h h;
        h[0] = (_Float16)f0.x; h[1] = (_Float16)f0.y;
        h[2] = (_Float16)f0.z; h[3] = (_Float16)f0.w;
        h[4] = (_Float16)f1.x; h[5] = (_Float16)f1.y;
        h[6] = (_Float16)f1.z; h[7] = (_Float16)f1.w;
        return h;
    };

    auto store_lds = [&](int b) {
        *(v8h*)&afrag[b][a_sub + 0][a_lane][a_j] = cvt8(aR0a, aR0b);
        *(v8h*)&afrag[b][a_sub + 4][a_lane][a_j] = cvt8(aR1a, aR1b);
        if (BL == 0) {
            *(v8h*)&bfrag[b][bnt_sub][bnt_lane][bnt_j] = cvt8(bRa, bRb);
        } else {
            const v8h h = cvt8(bRa, bRb);
            bfrag[b][bnn_sub][bnn_lane + 0][bnn_j] = h[0];
            bfrag[b][bnn_sub][bnn_lane + 1][bnn_j] = h[1];
            bfrag[b][bnn_sub][bnn_lane + 2][bnn_j] = h[2];
            bfrag[b][bnn_sub][bnn_lane + 3][bnn_j] = h[3];
            bfrag[b][bnn_sub][bnn_lane + 4][bnn_j] = h[4];
            bfrag[b][bnn_sub][bnn_lane + 5][bnn_j] = h[5];
            bfrag[b][bnn_sub][bnn_lane + 6][bnn_j] = h[6];
            bfrag[b][bnn_sub][bnn_lane + 7][bnn_j] = h[7];
        }
    };

    auto ldfrag = [](const _Float16* p) -> v16h {
        const v8h lo = *(const v8h*)p;
        const v8h hi = *(const v8h*)(p + 8);
        return __builtin_shufflevector(lo, hi,
                   0,1,2,3,4,5,6,7,8,9,10,11,12,13,14,15);
    };

    v8f acc[4] = {{}, {}, {}, {}};

    // prologue: stage tile 0
    load_regs(0);
    store_lds(0);
    __syncthreads();

    int buf = 0;
    for (int k0 = 0; k0 < K; k0 += 32) {
        const bool more = (k0 + 32) < K;
        if (more) load_regs(k0 + 32);            // VMEM issued under the WMMAs

        // Issue ALL fragment loads first (distinct registers) so the LDS
        // latency of later fragments hides under earlier WMMAs instead of
        // forcing a full s_wait_dscnt 0 before every WMMA.
        const v16h af  = ldfrag(afrag[buf][wave][lane]);
        const v16h bf0 = ldfrag(bfrag[buf][0][lane]);
        const v16h bf1 = ldfrag(bfrag[buf][1][lane]);
        const v16h bf2 = ldfrag(bfrag[buf][2][lane]);
        const v16h bf3 = ldfrag(bfrag[buf][3][lane]);

        acc[0] = __builtin_amdgcn_wmma_f32_16x16x32_f16(
                     false, af, false, bf0, (short)0, acc[0], false, false);
        acc[1] = __builtin_amdgcn_wmma_f32_16x16x32_f16(
                     false, af, false, bf1, (short)0, acc[1], false, false);
        acc[2] = __builtin_amdgcn_wmma_f32_16x16x32_f16(
                     false, af, false, bf2, (short)0, acc[2], false, false);
        acc[3] = __builtin_amdgcn_wmma_f32_16x16x32_f16(
                     false, af, false, bf3, (short)0, acc[3], false, false);

        if (more) store_lds(buf ^ 1);
        buf ^= 1;
        __syncthreads();
    }

    // ---- store C (VGPR v -> M = v + 8*(lane>=16), N = lane&15) ----
    const int mb = m0 + (wave << 4) + ((lane >> 4) << 3);
#pragma unroll
    for (int t = 0; t < 4; ++t) {
        const int cn = n0 + (t << 4) + (lane & 15);
        const float badd = bias ? bias[cn] : 0.0f;
#pragma unroll
        for (int v = 0; v < 8; ++v) {
            const int cm = mb + v;
            const float val = acc[t][v] + badd;
            if (CM == 0) {
                C[(long)blockIdx.z * sC + (long)cm * ldc + cn] = val;
            } else if (CM == 1) {
                const int b = cm >> 11, s = cm & (SEQ - 1);
                const int h = cn >> 6,  dd = cn & (DH - 1);
                C[(((long)(b * NH + h) * SEQ + s) << 6) + dd] = val;
            } else {
                const int b = blockIdx.z / NH, h = blockIdx.z % NH;
                C[((long)b * SEQ + cm) * DM + h * DH + cn] = val;
            }
        }
    }
}

// In-place scaled softmax over rows of length SEQ; one block per row.
__global__ __launch_bounds__(256)
void softmax_rows(float* __restrict__ W, float scale)
{
    __shared__ float red[256];
    const int tid = threadIdx.x;
    float* p = W + (long)blockIdx.x * SEQ + tid * 8;
    float4 x0 = *(const float4*)p;
    float4 x1 = *(const float4*)(p + 4);
    x0.x *= scale; x0.y *= scale; x0.z *= scale; x0.w *= scale;
    x1.x *= scale; x1.y *= scale; x1.z *= scale; x1.w *= scale;

    float m = fmaxf(fmaxf(fmaxf(x0.x, x0.y), fmaxf(x0.z, x0.w)),
                    fmaxf(fmaxf(x1.x, x1.y), fmaxf(x1.z, x1.w)));
    red[tid] = m;
    __syncthreads();
    for (int s = 128; s > 0; s >>= 1) {
        if (tid < s) red[tid] = fmaxf(red[tid], red[tid + s]);
        __syncthreads();
    }
    m = red[0];
    __syncthreads();

    x0.x = __expf(x0.x - m); x0.y = __expf(x0.y - m);
    x0.z = __expf(x0.z - m); x0.w = __expf(x0.w - m);
    x1.x = __expf(x1.x - m); x1.y = __expf(x1.y - m);
    x1.z = __expf(x1.z - m); x1.w = __expf(x1.w - m);

    red[tid] = x0.x + x0.y + x0.z + x0.w + x1.x + x1.y + x1.z + x1.w;
    __syncthreads();
    for (int s = 128; s > 0; s >>= 1) {
        if (tid < s) red[tid] += red[tid + s];
        __syncthreads();
    }
    const float inv = 1.0f / red[0];

    x0.x *= inv; x0.y *= inv; x0.z *= inv; x0.w *= inv;
    x1.x *= inv; x1.y *= inv; x1.z *= inv; x1.w *= inv;
    *(float4*)p       = x0;
    *(float4*)(p + 4) = x1;
}

extern "C" void kernel_launch(void* const* d_in, const int* in_sizes, int n_in,
                              void* d_out, int out_size, void* d_ws, size_t ws_size,
                              hipStream_t stream)
{
    (void)in_sizes; (void)n_in; (void)out_size; (void)ws_size;

    const float* Q  = (const float*)d_in[0];
    const float* Kx = (const float*)d_in[1];
    const float* V  = (const float*)d_in[2];
    const float* Wq = (const float*)d_in[3];
    const float* bq = (const float*)d_in[4];
    const float* Wk = (const float*)d_in[5];
    const float* bk = (const float*)d_in[6];
    const float* Wv = (const float*)d_in[7];
    const float* bv = (const float*)d_in[8];
    const float* Wo = (const float*)d_in[9];
    const float* bo = (const float*)d_in[10];

    float* out     = (float*)d_out;                       // [B,S,DM]
    float* weights = out + (long)BATCH * SEQ * DM;        // [B,H,S,S]

    const long hsz = (long)BATCH * NH * SEQ * DH;         // 3,145,728
    float* qh = (float*)d_ws;                             // [B,H,S,d]
    float* kh = qh + hsz;
    float* vh = kh + hsz;
    float* am = vh + hsz;                                 // merged attn [B,S,DM]

    const dim3 blk(256);

    // 1) Q/K/V projections: [4096x768] @ [768x768]^T -> head-split [B,H,S,d]
    const dim3 gp(DM / 64, (BATCH * SEQ) / 128, 1);
    gemm_wmma_f16<0, 1><<<gp, blk, 0, stream>>>(Q,  Wq, bq, qh, DM, DM, DM, 0, 0, 0, 0);
    gemm_wmma_f16<0, 1><<<gp, blk, 0, stream>>>(Kx, Wk, bk, kh, DM, DM, DM, 0, 0, 0, 0);
    gemm_wmma_f16<0, 1><<<gp, blk, 0, stream>>>(V,  Wv, bv, vh, DM, DM, DM, 0, 0, 0, 0);

    // 2) scores = q @ k^T per (b,h), written straight into the weights output
    const dim3 gs(SEQ / 64, SEQ / 128, BATCH * NH);
    gemm_wmma_f16<0, 0><<<gs, blk, 0, stream>>>(qh, kh, nullptr, weights,
                                                DH, DH, DH, SEQ,
                                                (long)SEQ * DH, (long)SEQ * DH,
                                                (long)SEQ * SEQ);

    // 3) in-place scaled softmax (scale = 1/sqrt(64))
    softmax_rows<<<BATCH * NH * SEQ, blk, 0, stream>>>(weights, 0.125f);

    // 4) attn = weights @ v per (b,h) -> merged [B,S,H*d]
    const dim3 ga(DH / 64, SEQ / 128, BATCH * NH);
    gemm_wmma_f16<1, 2><<<ga, blk, 0, stream>>>(weights, vh, nullptr, am,
                                                SEQ, SEQ, DH, 0,
                                                (long)SEQ * SEQ, (long)SEQ * DH, 0);

    // 5) out = attn @ Wo^T + bo
    gemm_wmma_f16<0, 0><<<gp, blk, 0, stream>>>(am, Wo, bo, out,
                                                DM, DM, DM, DM, 0, 0, 0);
}
